// KSparseFFTClassifier_80822694576249
// MI455X (gfx1250) — matrix operand order repaired and checked by reading, not compiled, not run on hardware
//
#include <hip/hip_runtime.h>
#include <hip/hip_bf16.h>
#include <math.h>

typedef __attribute__((ext_vector_type(16))) _Float16 v16h;
typedef __attribute__((ext_vector_type(8)))  float    v8f;

#define BS        4096
#define IN_DIM    2048
#define NOUT      16384
#define SLACKD    64
#define TOTAL_D   129
#define NT1       9            // GEMM1 N tiles (144 = 9*16, pads 129)
#define KS1       (IN_DIM/32)  // 64 k-steps in GEMM1
#define INV_SQRT_N   (1.0f/128.0f)
#define SQRT2_OVER_SQRTN 0.01104854345604f   // sqrt(2)/128
#define TWO_PI_OVER_N    3.834951969714103e-4f

// ---------- CDNA5 async global->LDS copy (ASYNCcnt), guarded ----------------
#ifndef __has_builtin
#define __has_builtin(x) 0
#endif
#if __has_builtin(__builtin_amdgcn_global_load_async_to_lds_b128) && \
    __has_builtin(__builtin_amdgcn_s_wait_asynccnt)
#define HAVE_ASYNC_LDS 1
#else
#define HAVE_ASYNC_LDS 0
#endif

#if HAVE_ASYNC_LDS
typedef int v4i_ __attribute__((vector_size(16)));
typedef __attribute__((address_space(1))) v4i_ gv4i;   // global 16B vector
typedef __attribute__((address_space(3))) v4i_ lv4i;   // LDS 16B vector
__device__ __forceinline__ void async_copy16(const void* g, void* l) {
    // per-lane 16-byte global->LDS async copy: global_load_async_to_lds_b128
    __builtin_amdgcn_global_load_async_to_lds_b128((gv4i*)g, (lv4i*)l, 0, 0);
}
__device__ __forceinline__ void async_wait0() {
    __builtin_amdgcn_s_wait_asynccnt(0);
}
#endif

// ---- fragment index helpers (wave32 WMMA f16 A/B frags, contiguous k = 16*half + e) ----
__device__ __forceinline__ int a_kr_from(int half, int e) {
    return half * 16 + e;
}
__device__ __forceinline__ void a_he_from_kr(int kr, int& half, int& e) {
    half = kr >> 4;
    e    = kr & 15;
}

// ---------------- kernel 1: build fused DFT+slack basis, B-frag layout -------
__global__ __launch_bounds__(256) void build_basis(_Float16* __restrict__ Bf,
                                                   const float* __restrict__ Ws) {
    int idx  = blockIdx.x * 256 + threadIdx.x;           // 2,097,152 total
    int e    = idx & 15;
    int lane = (idx >> 4) & 31;
    int ks   = (idx >> 9) & 3;
    int ntg  = idx >> 11;
    int n    = ntg * 16 + (lane & 15);
    int j    = ks * 32 + ((lane >> 4) << 4) + e;         // 0..127
    float val;
    if (j < 64) {
        int f = (j >> 1) + 1;                            // frequency 1..32
        int m = (f * n) & (NOUT - 1);                    // exact int phase reduction
        float th = (float)m * TWO_PI_OVER_N;
        float s, c;
        __sincosf(th, &s, &c);
        val = SQRT2_OVER_SQRTN * ((j & 1) ? s : c);
    } else {
        val = Ws[n * SLACKD + (j - 64)];
    }
    Bf[idx] = (_Float16)val;
}

// ---------------- kernel 2: x f32 -> f16 A-frag layout -----------------------
__global__ __launch_bounds__(256) void conv_x(const float* __restrict__ x,
                                              _Float16* __restrict__ xf) {
    int idx  = blockIdx.x * 256 + threadIdx.x;           // 8,388,608 total
    int e    = idx & 15;
    int lane = (idx >> 4) & 31;
    int ks   = (idx >> 9) & 63;
    int mt   = idx >> 15;
    int m    = mt * 16 + (lane & 15);
    int kr   = a_kr_from(lane >> 4, e);
    int k    = ks * 32 + kr;
    xf[idx] = (_Float16)x[(size_t)m * IN_DIM + k];
}

// ---------------- kernel 3: W_proj^T -> f16 B-frag layout, zero-pad to 144 ---
__global__ __launch_bounds__(256) void conv_w(const float* __restrict__ W,
                                              _Float16* __restrict__ wf) {
    int idx = blockIdx.x * 256 + threadIdx.x;
    if (idx >= NT1 * KS1 * 32 * 16) return;              // 294,912
    int e    = idx & 15;
    int lane = (idx >> 4) & 31;
    int ks   = (idx >> 9) & 63;
    int ntg  = idx >> 15;
    int n    = ntg * 16 + (lane & 15);
    int k    = ks * 32 + ((lane >> 4) << 4) + e;
    float val = (n < TOTAL_D) ? W[(size_t)n * IN_DIM + k] : 0.0f;
    wf[idx] = (_Float16)val;
}

// ---------------- kernel 4: GEMM1  h = x @ W^T + b ---------------------------
// 8 waves = 128 rows, N=144, K staged in double-buffered LDS; async overlap.
__global__ __launch_bounds__(256) void gemm1(const _Float16* __restrict__ xf,
                                             const _Float16* __restrict__ wf,
                                             const float* __restrict__ b_proj,
                                             _Float16* __restrict__ hf,
                                             float* __restrict__ dc) {
    __shared__ _Float16 ldsW[2][NT1 * 32 * 16];          // 2 x 9 KB
    const int tid = threadIdx.x, lane = tid & 31, wv = tid >> 5;
    const int mtile = blockIdx.x * 8 + wv;

    v8f zero = {0, 0, 0, 0, 0, 0, 0, 0};
    v8f acc[NT1];
#pragma unroll
    for (int t = 0; t < NT1; ++t) acc[t] = zero;

    const uint32_t* wf32 = (const uint32_t*)wf;

#if HAVE_ASYNC_LDS
    // prologue: async-stage slice ks=0 into buffer 0 (576 x 16B)
#pragma unroll
    for (int i = 0; i < 3; ++i) {
        int b = tid + i * 256;
        if (b < 576) {
            int t = b / 64, rem = b & 63;
            async_copy16((const char*)(wf32 + (size_t)(t * KS1 + 0) * 256) + rem * 16,
                         (char*)&ldsW[0][0] + (size_t)b * 16);
        }
    }
    async_wait0();
    __syncthreads();
    for (int ks = 0; ks < KS1; ++ks) {
        const int cur = ks & 1;
        if (ks + 1 < KS1) {                              // overlap: stage next slice
#pragma unroll
            for (int i = 0; i < 3; ++i) {
                int b = tid + i * 256;
                if (b < 576) {
                    int t = b / 64, rem = b & 63;
                    async_copy16((const char*)(wf32 + (size_t)(t * KS1 + ks + 1) * 256) + rem * 16,
                                 (char*)&ldsW[cur ^ 1][0] + (size_t)b * 16);
                }
            }
        }
        v16h a = *(const v16h*)(xf + ((size_t)(mtile * KS1 + ks) * 32 + lane) * 16);
        v16h barr[NT1];
#pragma unroll
        for (int t = 0; t < NT1; ++t)
            barr[t] = *(const v16h*)(&ldsW[cur][0] + (t * 32 + lane) * 16);
#pragma unroll
        for (int t = 0; t < NT1; ++t)
            acc[t] = __builtin_amdgcn_wmma_f32_16x16x32_f16(
                false, a, false, barr[t], (short)0, acc[t], false, false);
        async_wait0();
        __syncthreads();
    }
#else
    // prologue: synchronous stage of slice 0 into buffer 0 (2304 dwords)
    {
        uint32_t* lds32 = (uint32_t*)&ldsW[0][0];
#pragma unroll
        for (int i = 0; i < 9; ++i) {
            int di = tid + i * 256;
            int t = di >> 8, rem = di & 255;
            lds32[di] = wf32[(size_t)(t * KS1 + 0) * 256 + rem];
        }
    }
    __syncthreads();
    for (int ks = 0; ks < KS1; ++ks) {
        const int cur = ks & 1;
        uint32_t regs[9];
        if (ks + 1 < KS1) {                              // issue next-slice loads early
#pragma unroll
            for (int i = 0; i < 9; ++i) {
                int di = tid + i * 256;
                int t = di >> 8, rem = di & 255;
                regs[i] = wf32[(size_t)(t * KS1 + ks + 1) * 256 + rem];
            }
        }
        v16h a = *(const v16h*)(xf + ((size_t)(mtile * KS1 + ks) * 32 + lane) * 16);
        v16h barr[NT1];
#pragma unroll
        for (int t = 0; t < NT1; ++t)
            barr[t] = *(const v16h*)(&ldsW[cur][0] + (t * 32 + lane) * 16);
#pragma unroll
        for (int t = 0; t < NT1; ++t)
            acc[t] = __builtin_amdgcn_wmma_f32_16x16x32_f16(
                false, a, false, barr[t], (short)0, acc[t], false, false);
        if (ks + 1 < KS1) {
            uint32_t* lds32 = (uint32_t*)&ldsW[cur ^ 1][0];
#pragma unroll
            for (int i = 0; i < 9; ++i) lds32[tid + i * 256] = regs[i];
        }
        __syncthreads();
    }
#endif

    const int col = lane & 15, hh = lane >> 4;
#pragma unroll
    for (int t = 0; t < NT1; ++t) {
        int d = t * 16 + col;
        if (d >= TOTAL_D) continue;
        float bp = b_proj[d];
#pragma unroll
        for (int i = 0; i < 8; ++i) {
            int   m = mtile * 16 + hh * 8 + i;
            float v = acc[t][i] + bp;
            if (d == 0) {
                dc[m] = v;                               // keep DC exact in f32
            } else {
                int j = d - 1;
                int ks2 = j >> 5, kr = j & 31, h2, e2;
                a_he_from_kr(kr, h2, e2);
                int lane2 = h2 * 16 + (m & 15);
                hf[((size_t)((m >> 4) * 4 + ks2) * 32 + lane2) * 16 + e2] = (_Float16)v;
            }
        }
    }
}

// ---------------- kernel 5: GEMM2  out = h16 @ B16 + dc/sqrt(N) --------------
__global__ __launch_bounds__(256) void gemm2(const _Float16* __restrict__ hf,
                                             const _Float16* __restrict__ Bf,
                                             const float* __restrict__ dc,
                                             float* __restrict__ out) {
    __shared__ _Float16 ldsB[8 * 4 * 32 * 16];           // 32 KB
    const int tid = threadIdx.x, lane = tid & 31, wv = tid >> 5;
    const int mtile = blockIdx.x * 8 + wv;
    const int ntg0  = blockIdx.y * 8;

    // contiguous 32KB B-fragment slice for these 128 output columns
    const char* srcB = (const char*)(Bf + (size_t)ntg0 * 2048);
#if HAVE_ASYNC_LDS
#pragma unroll
    for (int i = 0; i < 8; ++i) {
        int b = tid + i * 256;
        async_copy16(srcB + (size_t)b * 16, (char*)ldsB + (size_t)b * 16);
    }
#else
    {
        const uint4* src = (const uint4*)srcB;
        uint4*       dst = (uint4*)ldsB;
#pragma unroll
        for (int i = 0; i < 8; ++i) dst[tid + i * 256] = src[tid + i * 256];
    }
#endif

    v16h a[4];
#pragma unroll
    for (int ks = 0; ks < 4; ++ks)
        a[ks] = *(const v16h*)(hf + ((size_t)(mtile * 4 + ks) * 32 + lane) * 16);

    const int hh = lane >> 4, col = lane & 15;
    float dcv[8];
#pragma unroll
    for (int i = 0; i < 8; ++i)
        dcv[i] = dc[mtile * 16 + hh * 8 + i] * INV_SQRT_N;

#if HAVE_ASYNC_LDS
    async_wait0();
#endif
    __syncthreads();

    v8f zero = {0, 0, 0, 0, 0, 0, 0, 0};
#pragma unroll
    for (int t = 0; t < 8; ++t) {
        v8f c = zero;
#pragma unroll
        for (int ks = 0; ks < 4; ++ks) {
            v16h b = *(const v16h*)(ldsB + ((t * 4 + ks) * 32 + lane) * 16);
            c = __builtin_amdgcn_wmma_f32_16x16x32_f16(
                false, a[ks], false, b, (short)0, c, false, false);
        }
        size_t n = (size_t)(ntg0 + t) * 16 + col;
#pragma unroll
        for (int i = 0; i < 8; ++i) {
            size_t m = (size_t)mtile * 16 + hh * 8 + i;
            __builtin_nontemporal_store(c[i] + dcv[i], out + m * NOUT + n);
        }
    }
}

// ---------------------------------------------------------------------------
extern "C" void kernel_launch(void* const* d_in, const int* in_sizes, int n_in,
                              void* d_out, int out_size, void* d_ws, size_t ws_size,
                              hipStream_t stream) {
    const float* x  = (const float*)d_in[0];   // 4096 x 2048
    const float* W  = (const float*)d_in[1];   // 129 x 2048
    const float* b  = (const float*)d_in[2];   // 129
    const float* Ws = (const float*)d_in[3];   // 16384 x 64
    float* out = (float*)d_out;                // 4096 x 16384

    char* ws = (char*)d_ws;                    // ~22.02 MB used
    _Float16* Bf = (_Float16*)(ws);                      //  4 MB  basis frags
    _Float16* xf = (_Float16*)(ws + (size_t)( 4 << 20)); // 16 MB  x frags
    _Float16* wf = (_Float16*)(ws + (size_t)(20 << 20)); // 576 KB W^T frags
    _Float16* hf = (_Float16*)(ws + (size_t)(21 << 20)); //  1 MB  h frags
    float*    dc = (float*)   (ws + (size_t)(22 << 20)); // 16 KB  DC column f32

    build_basis<<<(128 * NOUT) / 256, 256, 0, stream>>>(Bf, Ws);
    conv_x<<<(BS * IN_DIM) / 256, 256, 0, stream>>>(x, xf);
    conv_w<<<(NT1 * KS1 * 32 * 16 + 255) / 256, 256, 0, stream>>>(W, wf);
    gemm1<<<BS / 128, 256, 0, stream>>>(xf, wf, b, hf, dc);
    gemm2<<<dim3(BS / 128, NOUT / 128), 256, 0, stream>>>(hf, Bf, dc, out);
}
